// OddOneOutNet_10771777978673
// MI455X (gfx1250) — compile-verified, matrix operand order checked by reading, not run on hardware
//
#include <hip/hip_runtime.h>

// ---------------------------------------------------------------------------
// CDNA5 WMMA types / helpers
// ---------------------------------------------------------------------------
typedef __attribute__((ext_vector_type(16))) __bf16 v16bf;
typedef __attribute__((ext_vector_type(8)))  float  v8f;

union BF16x16 {
    unsigned short u[16];
    uint4          q[2];     // q[0] = halves 0..7, q[1] = halves 8..15
    v16bf          v;
};

static __device__ inline unsigned short f2bf(float f) {
    unsigned int u = __float_as_uint(f);
    unsigned int r = (u + 0x7fffu + ((u >> 16) & 1u)) >> 16;
    return (unsigned short)r;
}

// K index for packed-half slot j (0..15) of a 16x32 bf16 A / 32x16 B fragment.
// Per ISA 7.12.2: halves 0..7  <-> K = hi*8 + j
//                 halves 8..15 <-> K = 16 + hi*8 + (j-8)
static __device__ inline int kidx(int lane, int j) {
    int hi = (lane >> 4) & 1;
    return (j < 8) ? (hi * 8 + j) : (16 + hi * 8 + (j - 8));
}

static __device__ inline v8f wmma_bf16(v16bf a, v16bf b, v8f c) {
    return __builtin_amdgcn_wmma_f32_16x16x32_bf16(
        false, a, false, b, (short)0, c, false, false);
}

// ---------------------------------------------------------------------------
// Kernel 1: fused TinyCNN encoder. One workgroup (256 thr = 8 waves) / panel.
// Halo-padded bf16 LDS images; channel-innermost layouts so each A-fragment
// half-group is one aligned ds_load_b128. All gathers are BRANCHLESS: padding
// K positions hold finite garbage which the zeroed B-fragment rows annihilate.
// 2x2 maxpool folded into the D fragment.
// ---------------------------------------------------------------------------
__global__ __launch_bounds__(256) void encoder_kernel(
    const float* __restrict__ img,
    const float* __restrict__ w1, const float* __restrict__ b1,
    const float* __restrict__ w2, const float* __restrict__ b2,
    const float* __restrict__ w3, const float* __restrict__ b3,
    const float* __restrict__ fcw, const float* __restrict__ fcb,
    float* __restrict__ zcen)
{
    __shared__ unsigned short in_bfh[66 * 66];          // 8.7 KB, halo, 1 ch
    __shared__ unsigned short pool1h[34 * 34 * 8];      // 18.5 KB, [y][x][ci8]
    __shared__ unsigned short pool2h[18 * 18 * 16];     // 10.4 KB, [y][x][ci16]
    __shared__ float mean16[16];

    const int panel = blockIdx.x;
    const int tid   = threadIdx.x;
    const int lane  = tid & 31;
    const int wave  = tid >> 5;
    const int n     = lane & 15;   // output channel (B/D column)
    const int hi    = lane >> 4;

    // Zero all LDS images (halos + channel padding stay zero afterwards).
    {
        int* p0 = (int*)in_bfh;  // 2178 ints
        int* p1 = (int*)pool1h;  // 4624 ints
        int* p2 = (int*)pool2h;  // 2592 ints
        for (int i = tid; i < 2178; i += 256) p0[i] = 0;
        for (int i = tid; i < 4624; i += 256) p1[i] = 0;
        for (int i = tid; i < 2592; i += 256) p2[i] = 0;
        if (tid < 16) mean16[tid] = 0.0f;
    }
    __syncthreads();

    const float* src = img + (size_t)panel * 4096;
    for (int i = tid; i < 4096; i += 256)
        in_bfh[((i >> 6) + 1) * 66 + (i & 63) + 1] = f2bf(src[i]);
    __syncthreads();

    // ---------------- conv1: 1->8 ch, 64x64, K = tap (9, pad 32) ------------
    {
        BF16x16 B1t;
#pragma unroll
        for (int j = 0; j < 16; ++j) {
            int k = kidx(lane, j);
            float wv = (n < 8 && k < 9) ? w1[n * 9 + k] : 0.0f;
            B1t.u[j] = f2bf(wv);
        }
        float bias1 = (n < 8) ? b1[n] : 0.0f;

        for (int rp = wave; rp < 32; rp += 8) {       // pooled output row
            for (int xb = 0; xb < 4; ++xb) {          // 16-pixel x block
                int x = xb * 16 + (lane & 15);        // A row M -> pixel x
                float pl[4];
                for (int row = 0; row < 2; ++row) {
                    int y = rp * 2 + row;
                    int base = y * 66 + x;            // stored idx of tap (0,0)
                    BF16x16 A;
                    // lo lanes: slots 0..7 = taps 0..7.
                    // hi lanes: slot 0 = tap 8; slots 1..7 are garbage that
                    //           B1t zeroes (K 9..15).  Branchless.
                    A.u[0] = in_bfh[base + hi * 134];
                    A.u[1] = in_bfh[base + 1];
                    A.u[2] = in_bfh[base + 2];
                    A.u[3] = in_bfh[base + 66];
                    A.u[4] = in_bfh[base + 67];
                    A.u[5] = in_bfh[base + 68];
                    A.u[6] = in_bfh[base + 132];
                    A.u[7] = in_bfh[base + 133];
                    A.q[1] = uint4{0, 0, 0, 0};       // K 16..31 (padding)
                    v8f acc = {};
                    acc = wmma_bf16(A.v, B1t.v, acc);
                    // lane holds channel n, pixels x = xb*16 + hi*8 + r
#pragma unroll
                    for (int i = 0; i < 4; ++i) {
                        float m = fmaxf(acc[2 * i], acc[2 * i + 1]);   // horiz
                        pl[i] = (row == 0) ? m : fmaxf(pl[i], m);      // vert
                    }
                }
                if (n < 8) {
#pragma unroll
                    for (int i = 0; i < 4; ++i) {
                        float v = fmaxf(pl[i] + bias1, 0.0f);          // bias+relu
                        int xp = xb * 8 + hi * 4 + i;
                        pool1h[((rp + 1) * 34 + xp + 1) * 8 + n] = f2bf(v);
                    }
                }
            }
        }
    }
    __syncthreads();

    // ---------------- conv2: 8->12 ch, 32x32, K = tap*8+ci (72, pad 96) -----
    {
        BF16x16 B2t[3];
#pragma unroll
        for (int s = 0; s < 3; ++s)
#pragma unroll
            for (int j = 0; j < 16; ++j) {
                int k = s * 32 + kidx(lane, j);
                float wv = 0.0f;
                if (n < 12 && k < 72) { int tap = k >> 3, ci = k & 7; wv = w2[n * 72 + ci * 9 + tap]; }
                B2t[s].u[j] = f2bf(wv);
            }
        float bias2 = (n < 12) ? b2[n] : 0.0f;

        for (int rp = wave; rp < 16; rp += 8) {
            for (int xb = 0; xb < 2; ++xb) {
                int x = xb * 16 + (lane & 15);
                float pl[4];
                for (int row = 0; row < 2; ++row) {
                    int y = rp * 2 + row;
                    v8f acc = {};
#pragma unroll
                    for (int s = 0; s < 3; ++s) {
                        // halves 0..7 : tap 4s+hi ; halves 8..15 : tap 4s+2+hi
                        // clamp out-of-range taps to 8: loads valid (finite)
                        // garbage which B2t's zero rows annihilate.
                        int tapA = min(4 * s + hi, 8);
                        int tapB = min(4 * s + 2 + hi, 8);
                        BF16x16 A;
                        {
                            int ky = tapA / 3, kx = tapA % 3;
                            A.q[0] = *(const uint4*)&pool1h[((y + ky) * 34 + (x + kx)) * 8];
                        }
                        {
                            int ky = tapB / 3, kx = tapB % 3;
                            A.q[1] = *(const uint4*)&pool1h[((y + ky) * 34 + (x + kx)) * 8];
                        }
                        acc = wmma_bf16(A.v, B2t[s].v, acc);
                    }
#pragma unroll
                    for (int i = 0; i < 4; ++i) {
                        float m = fmaxf(acc[2 * i], acc[2 * i + 1]);
                        pl[i] = (row == 0) ? m : fmaxf(pl[i], m);
                    }
                }
                if (n < 12) {
#pragma unroll
                    for (int i = 0; i < 4; ++i) {
                        float v = fmaxf(pl[i] + bias2, 0.0f);
                        int xp = xb * 8 + hi * 4 + i;
                        pool2h[((rp + 1) * 18 + xp + 1) * 16 + n] = f2bf(v);
                    }
                }
            }
        }
    }
    __syncthreads();

    // ------------- conv3: 12->16 ch, 16x16, K = tap*16+ci (144, 5 steps) ----
    {
        BF16x16 B3t[5];
#pragma unroll
        for (int s = 0; s < 5; ++s)
#pragma unroll
            for (int j = 0; j < 16; ++j) {
                int k = s * 32 + kidx(lane, j);
                int tap = k >> 4, ci = k & 15;
                float wv = (tap < 9 && ci < 12) ? w3[n * 108 + ci * 9 + tap] : 0.0f;
                B3t[s].u[j] = f2bf(wv);
            }
        float bias3 = b3[n];

        for (int y = wave; y < 16; y += 8) {          // one 16-pixel row / tile
            int x = (lane & 15);
            v8f acc = {};
#pragma unroll
            for (int s = 0; s < 5; ++s) {
                // halves 0..7: tap 2s, ci hi*8+{0..7}; halves 8..15: tap 2s+1.
                // Clamp tap 9 (s=4 group1) to 8: B3t[4] zeroes those K rows.
                int t0 = 2 * s;
                int t1 = (2 * s + 1 > 8) ? 8 : (2 * s + 1);
                BF16x16 A;
                {
                    int ky = t0 / 3, kx = t0 % 3;
                    A.q[0] = *(const uint4*)&pool2h[((y + ky) * 18 + (x + kx)) * 16 + hi * 8];
                }
                {
                    int ky = t1 / 3, kx = t1 % 3;
                    A.q[1] = *(const uint4*)&pool2h[((y + ky) * 18 + (x + kx)) * 16 + hi * 8];
                }
                acc = wmma_bf16(A.v, B3t[s].v, acc);
            }
            float partial = 0.0f;
#pragma unroll
            for (int r = 0; r < 8; ++r) partial += fmaxf(acc[r] + bias3, 0.0f);  // relu
            atomicAdd(&mean16[n], partial);            // ds_add_f32 (GAP)
        }
    }
    __syncthreads();

    // ---------------- GAP -> FC (16 -> 20) ----------------------------------
    if (tid < 20) {
        float s = fcb[tid];
        for (int c = 0; c < 16; ++c) s += fcw[tid * 16 + c] * (mean16[c] * (1.0f / 256.0f));
        zcen[(size_t)panel * 20 + tid] = s;
    }
}

// ---------------------------------------------------------------------------
// Kernel 2: everything after the encoder. One 64-thread WG per batch element.
// ---------------------------------------------------------------------------
__global__ __launch_bounds__(64) void stage2_kernel(
    const float* __restrict__ zcen, const float* __restrict__ meta,
    const float* __restrict__ m1w, const float* __restrict__ m1b,
    const float* __restrict__ m2w, const float* __restrict__ m2b,
    const float* __restrict__ p1w, const float* __restrict__ p1b,
    const float* __restrict__ p2w, const float* __restrict__ p2b,
    const float* __restrict__ centers, const float* __restrict__ archw, const float* __restrict__ archb,
    const float* __restrict__ ipw, const float* __restrict__ ipb,
    const float* __restrict__ opw, const float* __restrict__ opb,
    const float* __restrict__ ln1g, const float* __restrict__ ln1b,
    const float* __restrict__ ff1w, const float* __restrict__ ff1b,
    const float* __restrict__ ff2w, const float* __restrict__ ff2b,
    const float* __restrict__ ln2g, const float* __restrict__ ln2b,
    const float* __restrict__ r1w, const float* __restrict__ r1b,
    const float* __restrict__ r2w, const float* __restrict__ r2b,
    const float* __restrict__ s1w, const float* __restrict__ s1b,
    const float* __restrict__ s2w, const float* __restrict__ s2b,
    float* __restrict__ out)
{
    const int b = blockIdx.x;
    const int tid = threadIdx.x;

    __shared__ float zc[5][20], mt[5][16];
    __shared__ float h1[5][12], zm[5][12];
    __shared__ float cat[5][32], ph[5][24], h0[5][24];
    __shared__ float qv[5][6];
    __shared__ float cen[5][24], coordS[5][24], x0[5][24];
    __shared__ float qkv[5][72];
    __shared__ float sc[2][5][5];
    __shared__ float atmp[5][24], attn[5][24], x1[5][24];
    __shared__ float ffh[5][48], hset[5][24], mo[5][24];
    __shared__ float relS[25][10], rmean[5][10], rmax[5][10];
    __shared__ float sf[5][146], sh[5][40];

    for (int i = tid; i < 100; i += 64) zc[i / 20][i % 20] = zcen[(size_t)b * 100 + i];
    for (int i = tid; i < 80;  i += 64) mt[i / 16][i % 16] = meta[(size_t)b * 80 + i];
    __syncthreads();

    // meta MLP: 16 -> relu 12 -> 12
    for (int i = tid; i < 60; i += 64) {
        int it = i / 12, o = i % 12;
        float s = m1b[o];
        for (int c = 0; c < 16; ++c) s += m1w[o * 16 + c] * mt[it][c];
        h1[it][o] = fmaxf(s, 0.0f);
    }
    __syncthreads();
    for (int i = tid; i < 60; i += 64) {
        int it = i / 12, o = i % 12;
        float s = m2b[o];
        for (int c = 0; c < 12; ++c) s += m2w[o * 12 + c] * h1[it][c];
        zm[it][o] = s;
    }
    __syncthreads();

    // concat + projection: 32 -> relu 24 -> 24
    for (int i = tid; i < 160; i += 64) {
        int it = i / 32, d = i % 32;
        cat[it][d] = (d < 20) ? zc[it][d] : zm[it][d - 20];
    }
    __syncthreads();
    for (int i = tid; i < 120; i += 64) {
        int it = i / 24, o = i % 24;
        float s = p1b[o];
        for (int c = 0; c < 32; ++c) s += p1w[o * 32 + c] * cat[it][c];
        ph[it][o] = fmaxf(s, 0.0f);
    }
    __syncthreads();
    for (int i = tid; i < 120; i += 64) {
        int it = i / 24, o = i % 24;
        float s = p2b[o];
        for (int c = 0; c < 24; ++c) s += p2w[o * 24 + c] * ph[it][c];
        h0[it][o] = s;
    }
    __syncthreads();

    // archetype routing: softmax(h0 @ arch_w.T + b)
    for (int i = tid; i < 30; i += 64) {
        int it = i / 6, k = i % 6;
        float s = archb[k];
        for (int d = 0; d < 24; ++d) s += archw[k * 24 + d] * h0[it][d];
        qv[it][k] = s;
    }
    __syncthreads();
    if (tid < 5) {
        float mx = qv[tid][0];
        for (int k = 1; k < 6; ++k) mx = fmaxf(mx, qv[tid][k]);
        float sum = 0.0f;
        for (int k = 0; k < 6; ++k) { float e = __expf(qv[tid][k] - mx); qv[tid][k] = e; sum += e; }
        float inv = 1.0f / sum;
        for (int k = 0; k < 6; ++k) qv[tid][k] *= inv;
    }
    __syncthreads();
    for (int i = tid; i < 120; i += 64) {
        int it = i / 24, d = i % 24;
        float c = 0.0f;
        for (int k = 0; k < 6; ++k) c += qv[it][k] * centers[k * 24 + d];
        cen[it][d]    = c;
        coordS[it][d] = h0[it][d] - c;
        x0[it][d]     = h0[it][d] + c;
    }
    __syncthreads();

    // attention: qkv projection (72 units/item)
    for (int i = tid; i < 360; i += 64) {
        int it = i / 72, o = i % 72;
        float s = ipb[o];
        for (int d = 0; d < 24; ++d) s += ipw[o * 24 + d] * x0[it][d];
        qkv[it][o] = s;
    }
    __syncthreads();
    for (int i = tid; i < 50; i += 64) {
        int h = i / 25, qi = (i % 25) / 5, ki = i % 5;
        float s = 0.0f;
        for (int d = 0; d < 12; ++d) s += qkv[qi][h * 12 + d] * qkv[ki][24 + h * 12 + d];
        sc[h][qi][ki] = s * 0.28867513459f;   // 1/sqrt(12)
    }
    __syncthreads();
    if (tid < 10) {
        int h = tid / 5, qi = tid % 5;
        float mx = sc[h][qi][0];
        for (int k = 1; k < 5; ++k) mx = fmaxf(mx, sc[h][qi][k]);
        float sum = 0.0f;
        for (int k = 0; k < 5; ++k) { float e = __expf(sc[h][qi][k] - mx); sc[h][qi][k] = e; sum += e; }
        float inv = 1.0f / sum;
        for (int k = 0; k < 5; ++k) sc[h][qi][k] *= inv;
    }
    __syncthreads();
    for (int i = tid; i < 120; i += 64) {
        int qi = i / 24, d24 = i % 24, h = d24 / 12, d = d24 % 12;
        float s = 0.0f;
        for (int k = 0; k < 5; ++k) s += sc[h][qi][k] * qkv[k][48 + h * 12 + d];
        atmp[qi][d24] = s;
    }
    __syncthreads();
    for (int i = tid; i < 120; i += 64) {
        int it = i / 24, o = i % 24;
        float s = opb[o];
        for (int d = 0; d < 24; ++d) s += opw[o * 24 + d] * atmp[it][d];
        attn[it][o] = s;
    }
    __syncthreads();
    if (tid < 5) {  // x1 = LN(x0 + attn)
        float m = 0.0f;
        for (int d = 0; d < 24; ++d) m += x0[tid][d] + attn[tid][d];
        m *= (1.0f / 24.0f);
        float v = 0.0f;
        for (int d = 0; d < 24; ++d) { float t = x0[tid][d] + attn[tid][d] - m; v += t * t; }
        float inv = rsqrtf(v * (1.0f / 24.0f) + 1e-5f);
        for (int d = 0; d < 24; ++d)
            x1[tid][d] = (x0[tid][d] + attn[tid][d] - m) * inv * ln1g[d] + ln1b[d];
    }
    __syncthreads();

    // FFN: 24 -> relu 48 -> 24 ; hset = LN(x1 + ff)
    for (int i = tid; i < 240; i += 64) {
        int it = i / 48, o = i % 48;
        float s = ff1b[o];
        for (int d = 0; d < 24; ++d) s += ff1w[o * 24 + d] * x1[it][d];
        ffh[it][o] = fmaxf(s, 0.0f);
    }
    __syncthreads();
    for (int i = tid; i < 120; i += 64) {
        int it = i / 24, o = i % 24;
        float s = ff2b[o];
        for (int c = 0; c < 48; ++c) s += ff2w[o * 48 + c] * ffh[it][c];
        atmp[it][o] = x1[it][o] + s;   // reuse atmp as pre-LN buffer
    }
    __syncthreads();
    if (tid < 5) {
        float m = 0.0f;
        for (int d = 0; d < 24; ++d) m += atmp[tid][d];
        m *= (1.0f / 24.0f);
        float v = 0.0f;
        for (int d = 0; d < 24; ++d) { float t = atmp[tid][d] - m; v += t * t; }
        float inv = rsqrtf(v * (1.0f / 24.0f) + 1e-5f);
        for (int d = 0; d < 24; ++d)
            hset[tid][d] = (atmp[tid][d] - m) * inv * ln2g[d] + ln2b[d];
    }
    __syncthreads();
    for (int i = tid; i < 120; i += 64) {
        int it = i / 24, d = i % 24;
        float s = 0.0f;
        for (int j = 0; j < 5; ++j) s += hset[j][d];
        mo[it][d] = (s - hset[it][d]) * 0.25f;
    }
    __syncthreads();

    // pairwise relation MLP: feat(74) -> relu 10 -> relu 10
    if (tid < 25) {
        int pi = tid / 5, pj = tid % 5;
        float qov = 0.0f, d2 = 0.0f;
        for (int k = 0; k < 6; ++k) qov += qv[pi][k] * qv[pj][k];
        for (int d = 0; d < 24; ++d) { float c = coordS[pi][d] - coordS[pj][d]; d2 += c * c; }
        float dist = (d2 > 0.0f) ? sqrtf(d2) : 0.0f;
        float r1[10];
        for (int o = 0; o < 10; ++o) {
            const float* w = r1w + o * 74;
            float s = r1b[o];
            for (int d = 0; d < 24; ++d) s += w[d]      * hset[pi][d];
            for (int d = 0; d < 24; ++d) s += w[24 + d] * hset[pj][d];
            for (int d = 0; d < 24; ++d) s += w[48 + d] * fabsf(coordS[pi][d] - coordS[pj][d]);
            s += w[72] * qov + w[73] * dist;
            r1[o] = fmaxf(s, 0.0f);
        }
        for (int o = 0; o < 10; ++o) {
            float s = r2b[o];
            for (int c = 0; c < 10; ++c) s += r2w[o * 10 + c] * r1[c];
            relS[tid][o] = fmaxf(s, 0.0f);
        }
    }
    __syncthreads();
    for (int i = tid; i < 50; i += 64) {
        int pi = i / 10, o = i % 10;
        float s = 0.0f, mx = -1e9f;
        for (int pj = 0; pj < 5; ++pj) {
            if (pj == pi) continue;
            float v = relS[pi * 5 + pj][o];
            s += v; mx = fmaxf(mx, v);
        }
        rmean[pi][o] = s * 0.25f;
        rmax[pi][o]  = mx;
    }
    __syncthreads();

    // scorer feature (146) + MLP 146 -> relu 40 -> 1
    for (int i = tid; i < 730; i += 64) {
        int it = i / 146, t = i % 146;
        float v;
        if      (t < 24)  v = h0[it][t];
        else if (t < 48)  v = hset[it][t - 24];
        else if (t < 72)  v = cen[it][t - 48];
        else if (t < 96)  v = coordS[it][t - 72];
        else if (t < 120) v = fabsf(hset[it][t - 96] - mo[it][t - 96]);
        else if (t < 130) v = rmean[it][t - 120];
        else if (t < 140) v = rmax[it][t - 130];
        else              v = qv[it][t - 140];
        sf[it][t] = v;
    }
    __syncthreads();
    for (int i = tid; i < 200; i += 64) {
        int it = i / 40, o = i % 40;
        float s = s1b[o];
        for (int t = 0; t < 146; ++t) s += s1w[o * 146 + t] * sf[it][t];
        sh[it][o] = fmaxf(s, 0.0f);
    }
    __syncthreads();
    if (tid < 5) {
        float s = s2b[0];
        for (int o = 0; o < 40; ++o) s += s2w[o] * sh[tid][o];
        out[(size_t)b * 5 + tid] = s;                 // logits
    }
    for (int i = tid; i < 30; i += 64)
        out[5120 + (size_t)b * 30 + i] = qv[i / 6][i % 6];  // q
}

// ---------------------------------------------------------------------------
extern "C" void kernel_launch(void* const* d_in, const int* in_sizes, int n_in,
                              void* d_out, int out_size, void* d_ws, size_t ws_size,
                              hipStream_t stream) {
    (void)in_sizes; (void)n_in; (void)out_size; (void)ws_size;
    const float* centered = (const float*)d_in[0];
    const float* meta     = (const float*)d_in[1];
    const float* w1  = (const float*)d_in[2];  const float* b1  = (const float*)d_in[3];
    const float* w2  = (const float*)d_in[4];  const float* b2  = (const float*)d_in[5];
    const float* w3  = (const float*)d_in[6];  const float* b3  = (const float*)d_in[7];
    const float* fcw = (const float*)d_in[8];  const float* fcb = (const float*)d_in[9];
    const float* m1w = (const float*)d_in[10]; const float* m1b = (const float*)d_in[11];
    const float* m2w = (const float*)d_in[12]; const float* m2b = (const float*)d_in[13];
    const float* p1w = (const float*)d_in[14]; const float* p1b = (const float*)d_in[15];
    const float* p2w = (const float*)d_in[16]; const float* p2b = (const float*)d_in[17];
    const float* ctr = (const float*)d_in[18];
    const float* aw  = (const float*)d_in[19]; const float* ab  = (const float*)d_in[20];
    const float* ipw = (const float*)d_in[21]; const float* ipb = (const float*)d_in[22];
    const float* opw = (const float*)d_in[23]; const float* opb = (const float*)d_in[24];
    const float* l1g = (const float*)d_in[25]; const float* l1b = (const float*)d_in[26];
    const float* f1w = (const float*)d_in[27]; const float* f1b = (const float*)d_in[28];
    const float* f2w = (const float*)d_in[29]; const float* f2b = (const float*)d_in[30];
    const float* l2g = (const float*)d_in[31]; const float* l2b = (const float*)d_in[32];
    const float* r1w = (const float*)d_in[33]; const float* r1b = (const float*)d_in[34];
    const float* r2w = (const float*)d_in[35]; const float* r2b = (const float*)d_in[36];
    const float* s1w = (const float*)d_in[37]; const float* s1b = (const float*)d_in[38];
    const float* s2w = (const float*)d_in[39]; const float* s2b = (const float*)d_in[40];

    float* zcen = (float*)d_ws;   // 1024*5*20 floats

    encoder_kernel<<<5120, 256, 0, stream>>>(centered, w1, b1, w2, b2, w3, b3,
                                             fcw, fcb, zcen);
    stage2_kernel<<<1024, 64, 0, stream>>>(zcen, meta,
                                           m1w, m1b, m2w, m2b, p1w, p1b, p2w, p2b,
                                           ctr, aw, ab, ipw, ipb, opw, opb,
                                           l1g, l1b, f1w, f1b, f2w, f2b, l2g, l2b,
                                           r1w, r1b, r2w, r2b, s1w, s1b, s2w, s2b,
                                           (float*)d_out);
}
